// MultiBoxLoss_7138235646486
// MI455X (gfx1250) — compile-verified, matrix operand order checked by reading, not run on hardware
//
#include <hip/hip_runtime.h>
#include <math.h>

// ---------------- CDNA5 (gfx1250, wave32) SSD MultiBoxLoss ----------------
// Pipeline: init -> best-prior argmax -> match/encode/SmoothL1 ->
//           CE (async-LDS stream + WMMA exp-rowsum) + histogram ->
//           top-k scan -> select -> final.
// Bandwidth-bound (~165MB @ 23.3TB/s). conf_data streamed to LDS with
// global_load_async_to_lds_b128 (ASYNCcnt); softmax row-sum on the matrix
// pipe (v_wmma_f32_16x16x32_f16, f16 exp values, f32 accumulate).

typedef __attribute__((ext_vector_type(16))) _Float16 v16h;
typedef __attribute__((ext_vector_type(8)))  float    v8f;

#define POS_T 0.5f
#define NEG_T 0.4f
#define VAR0  0.1f
#define VAR1  0.2f
#define NBINS 65536

struct Scal {
    int   pos_count;
    int   cand_count;
    float pos_loss;
    int   tb;        // threshold bin (0x10000 => select nothing)
    int   rneed;     // how many to take from boundary bin
    int   ticket;
    float keep_sum;
    int   keep_cnt;
    int   num_neg;
};

// low 32 bits of a flat LDS pointer == workgroup-relative LDS byte offset
__device__ __forceinline__ unsigned lds_off(const void* p) {
    return (unsigned)(unsigned long long)p;
}

// ---------------- K0: zero accumulators + histogram ----------------
__global__ void k_init(int* __restrict__ hist, Scal* __restrict__ sc) {
    int i = blockIdx.x * blockDim.x + threadIdx.x;
    if (i < NBINS) hist[i] = 0;
    if (i == 0) {
        sc->pos_count = 0; sc->cand_count = 0; sc->pos_loss = 0.f;
        sc->keep_sum = 0.f; sc->keep_cnt = 0; sc->ticket = 0;
        sc->tb = 0x10000; sc->rneed = 0; sc->num_neg = 0;
    }
}

// ---------------- K1: best prior per (b,g): argmax_p IoU ----------------
__global__ void k_best_prior(const float* __restrict__ gt,
                             const float* __restrict__ priors,
                             int P, int* __restrict__ bpi) {
    int bg = blockIdx.x;                     // b*G + g
    const float* box = gt + (size_t)bg * 4;
    float gx1 = box[0], gy1 = box[1], gx2 = box[2], gy2 = box[3];
    float garea = (gx2 - gx1) * (gy2 - gy1);

    float bestOv = -1.f;
    int   bestIdx = 0x7fffffff;
    for (int p = threadIdx.x; p < P; p += blockDim.x) {
        float4 pr = ((const float4*)priors)[p];
        float px1 = pr.x - pr.z * 0.5f, py1 = pr.y - pr.w * 0.5f;
        float px2 = pr.x + pr.z * 0.5f, py2 = pr.y + pr.w * 0.5f;
        float ix = fmaxf(fminf(gx2, px2) - fmaxf(gx1, px1), 0.f);
        float iy = fmaxf(fminf(gy2, py2) - fmaxf(gy1, py1), 0.f);
        float inter = ix * iy;
        float ov = inter / (garea + (px2 - px1) * (py2 - py1) - inter);
        if (ov > bestOv) { bestOv = ov; bestIdx = p; }   // first-max tie rule
    }
    __shared__ float sOv[256];
    __shared__ int   sIdx[256];
    sOv[threadIdx.x] = bestOv; sIdx[threadIdx.x] = bestIdx;
    __syncthreads();
    for (int s = 128; s > 0; s >>= 1) {
        if ((int)threadIdx.x < s) {
            float o2 = sOv[threadIdx.x + s]; int i2 = sIdx[threadIdx.x + s];
            if (o2 > sOv[threadIdx.x] ||
                (o2 == sOv[threadIdx.x] && i2 < sIdx[threadIdx.x])) {
                sOv[threadIdx.x] = o2; sIdx[threadIdx.x] = i2;
            }
        }
        __syncthreads();
    }
    if (threadIdx.x == 0) bpi[bg] = sIdx[0];
}

// ---------------- K2: match per (b,p), conf_t, SmoothL1 over positives ----
__global__ void k_match(const float* __restrict__ loc,
                        const float* __restrict__ gt,
                        const int*   __restrict__ labels,
                        const float* __restrict__ priors,
                        const int*   __restrict__ bpi,
                        int G, int P,
                        int* __restrict__ conf_t, Scal* __restrict__ sc) {
    int blocksPerB = (P + 255) >> 8;
    int b = blockIdx.x / blocksPerB;
    int p = (blockIdx.x % blocksPerB) * 256 + threadIdx.x;

    __shared__ float sg[64 * 4];
    __shared__ int   sl[64];
    __shared__ int   sb[64];
    __shared__ int   sPos, sCand;
    __shared__ float sLoss;
    if (threadIdx.x == 0) { sPos = 0; sCand = 0; sLoss = 0.f; }
    if ((int)threadIdx.x < G) {
        int g = threadIdx.x;
        const float* bx = gt + ((size_t)b * G + g) * 4;
        sg[g * 4 + 0] = bx[0]; sg[g * 4 + 1] = bx[1];
        sg[g * 4 + 2] = bx[2]; sg[g * 4 + 3] = bx[3];
        sl[g] = labels[b * G + g];
        sb[g] = bpi[b * G + g];
    }
    __syncthreads();

    int myPos = 0, myCand = 0;
    float myLoss = 0.f;
    if (p < P) {
        float4 pr = ((const float4*)priors)[p];
        float px1 = pr.x - pr.z * 0.5f, py1 = pr.y - pr.w * 0.5f;
        float px2 = pr.x + pr.z * 0.5f, py2 = pr.y + pr.w * 0.5f;
        float parea = (px2 - px1) * (py2 - py1);

        float bestOv = -1.f; int bi = 0;
        for (int g = 0; g < G; ++g) {
            float gx1 = sg[g*4], gy1 = sg[g*4+1], gx2 = sg[g*4+2], gy2 = sg[g*4+3];
            float ix = fmaxf(fminf(gx2, px2) - fmaxf(gx1, px1), 0.f);
            float iy = fmaxf(fminf(gy2, py2) - fmaxf(gy1, py1), 0.f);
            float inter = ix * iy;
            float ov = inter / ((gx2-gx1)*(gy2-gy1) + parea - inter);
            if (ov > bestOv) { bestOv = ov; bi = g; }
        }
        // force-match: scatter semantics => later g wins on duplicates
        for (int g = 0; g < G; ++g)
            if (sb[g] == p) { bestOv = 2.0f; bi = g; }

        int c;
        if (bestOv < NEG_T)      c = 0;     // negative
        else if (bestOv < POS_T) c = -1;    // neutral
        else                     c = sl[bi] + 1;
        conf_t[(size_t)b * P + p] = c;

        if (c > 0) {
            float mx1 = sg[bi*4], my1 = sg[bi*4+1], mx2 = sg[bi*4+2], my2 = sg[bi*4+3];
            float gcx = ((mx1 + mx2) * 0.5f - pr.x) / (VAR0 * pr.z);
            float gcy = ((my1 + my2) * 0.5f - pr.y) / (VAR0 * pr.w);
            float gw  = __logf((mx2 - mx1) / pr.z) / VAR1;
            float gh  = __logf((my2 - my1) / pr.w) / VAR1;
            float4 ld = ((const float4*)loc)[(size_t)b * P + p];
            float d[4] = { ld.x - gcx, ld.y - gcy, ld.z - gw, ld.w - gh };
            float s = 0.f;
#pragma unroll
            for (int k = 0; k < 4; ++k) {
                float a = fabsf(d[k]);
                s += (a < 1.f) ? 0.5f * d[k] * d[k] : a - 0.5f;
            }
            myLoss = s; myPos = 1;
        } else if (c == 0) {
            myCand = 1;
        }
    }
    if (myPos) { atomicAdd(&sPos, 1); atomicAdd(&sLoss, myLoss); }
    if (myCand) atomicAdd(&sCand, 1);
    __syncthreads();
    if (threadIdx.x == 0) {
        if (sPos)  atomicAdd(&sc->pos_count, sPos);
        if (sCand) atomicAdd(&sc->cand_count, sCand);
        if (sLoss != 0.f) atomicAdd(&sc->pos_loss, sLoss);
    }
}

// ---------------- K3: CE per row; async global->LDS stream + WMMA --------
// One wave handles 16 rows (16*C contiguous floats, 5184B for C=81),
// streamed with global_load_async_to_lds_b128 (ASYNCcnt-tracked, no VGPR
// round trip). exp row-sum via v_wmma_f32_16x16x32_f16; A-fragment packing
// per ISA 16-bit A layout (lanes 0-15: K in {0..7,16..23}; lanes 16-31: +8),
// index-clamped so the per-element loads stay branch-free.
__global__ void __launch_bounds__(256)
k_ce(const float* __restrict__ conf, const int* __restrict__ conf_t,
     int BP, int C, float* __restrict__ ce_out, int* __restrict__ hist) {
    const int wave = threadIdx.x >> 5;
    const int lane = threadIdx.x & 31;
    __shared__ __align__(16) float tile[8][16 * 96];

    int numTiles = (BP + 15) >> 4;
    int tileIdx  = blockIdx.x * 8 + wave;
    bool active  = tileIdx < numTiles;
    int rowBase  = tileIdx * 16;
    long base    = (long)rowBase * C;
    int  nflt    = 16 * C;

    if (active) {
        bool full = (rowBase + 16) <= BP;
        if (full) {
            const float4* src = (const float4*)(conf + base); // 64*C bytes aligned
            __builtin_prefetch(conf + base + 8 * nflt, 0, 1); // next-block stream hint
            unsigned dbase = lds_off(&tile[wave][0]);
            int nv = nflt >> 2;
            for (int i = lane; i < nv; i += 32) {
                unsigned d = dbase + (unsigned)i * 16u;
                const float4* g = src + i;
                asm volatile("global_load_async_to_lds_b128 %0, %1, off"
                             :: "v"(d), "v"(g) : "memory");
            }
            asm volatile("s_wait_asynccnt 0" ::: "memory");
        } else {
            long tot = (long)BP * C;
            for (int i = lane; i < nflt; i += 32) {
                long g = base + i;
                tile[wave][i] = (g < tot) ? conf[g] : 0.f;
            }
        }
    }
    __syncthreads();

    if (active) {
        int   m    = lane & 15;
        int   half = lane >> 4;
        const float* row = &tile[wave][m * C];

        // row max: split across lane halves, 4-wide unrolled, xor-16 combine
        int split = (C + 1) >> 1;
        int k0 = half ? split : 0;
        int k1 = half ? C : split;
        float mx = -1e30f;
        int k = k0;
        for (; k + 3 < k1; k += 4) {
            float a0 = fmaxf(row[k],     row[k + 1]);
            float a1 = fmaxf(row[k + 2], row[k + 3]);
            mx = fmaxf(mx, fmaxf(a0, a1));
        }
        for (; k < k1; ++k) mx = fmaxf(mx, row[k]);
        mx = fmaxf(mx, __shfl_xor(mx, 16, 32));

        // sum_k exp(x-mx) via WMMA: A = exp fragment (f16), B = ones, C = f32
        v16h bones;
#pragma unroll
        for (int j = 0; j < 16; ++j) bones[j] = (_Float16)1.0f;
        v8f acc = {};
        int laneOff = half ? 8 : 0;
        int Cm1 = C - 1;
#pragma unroll
        for (int c = 0; c < 3; ++c) {           // 3 x K=32 covers C <= 96
            v16h a;
#pragma unroll
            for (int j = 0; j < 16; ++j) {
                int klocal = (j < 8) ? j : (j + 8);
                int kk = c * 32 + klocal + laneOff;
                int kc = kk < Cm1 ? kk : Cm1;            // clamp: load always valid
                float e = __expf(row[kc] - mx);          // unconditional ds_load
                a[j] = (_Float16)((kk < C) ? e : 0.f);   // v_cndmask, no branch
            }
            acc = __builtin_amdgcn_wmma_f32_16x16x32_f16(
                false, a, false, bones, (short)0, acc, false, false);
        }
        // D layout: row m sum lives in VGPR (m&7), lane half (m<8 ? lo : hi)
        int   idx     = m & 7;
        float mine    = acc[idx];
        float partner = __shfl_xor(mine, 16, 32);
        float ssum    = ((m < 8) == (half == 0)) ? mine : partner;

        if (half == 0) {
            int r = rowBase + m;
            if (r < BP) {
                int ct = conf_t[r];
                int t  = ct > 0 ? ct : 0;                 // clip(conf_t, 0)
                float ce = -(row[t] - mx - __logf(ssum)); // -log_softmax[t]
                ce_out[r] = ce;
                if (ct == 0) {                            // hard-neg candidate
                    unsigned bits = __float_as_uint(ce);  // ce>=0: monotonic bits
                    atomicAdd(&hist[bits >> 16], 1);
                }
            }
        }
    }
}

// ---------------- K4: descending histogram scan -> threshold bin ---------
__global__ void k_scan(const int* __restrict__ hist, Scal* __restrict__ sc) {
    __shared__ int chunk[256];
    int t = threadIdx.x;
    int s = 0;
    for (int j = 0; j < 256; ++j) s += hist[65535 - (t * 256 + j)];
    chunk[t] = s;
    __syncthreads();
    if (t == 0) {
        int np = sc->pos_count;
        int nn = 3 * np;
        int cc = sc->cand_count;
        int num_neg = nn < cc ? nn : cc;
        sc->num_neg = num_neg;
        int tb = 0x10000, rneed = 0;
        if (num_neg > 0) {
            int cum = 0, c = 0;
            for (; c < 256; ++c) {
                if (cum + chunk[c] >= num_neg) break;
                cum += chunk[c];
            }
            if (c < 256) {
                for (int j = 0; j < 256; ++j) {
                    int bin = 65535 - (c * 256 + j);
                    cum += hist[bin];
                    if (cum >= num_neg) {
                        tb = bin;
                        rneed = hist[bin] - (cum - num_neg);
                        break;
                    }
                }
            }
        }
        sc->tb = tb; sc->rneed = rneed; sc->ticket = 0;
    }
}

// ---------------- K5: keep = pos | (cand above threshold) ----------------
__global__ void k_select(const int* __restrict__ conf_t,
                         const float* __restrict__ ce,
                         int BP, Scal* __restrict__ sc) {
    __shared__ float sSum;
    __shared__ int   sCnt;
    if (threadIdx.x == 0) { sSum = 0.f; sCnt = 0; }
    __syncthreads();
    int i = blockIdx.x * blockDim.x + threadIdx.x;
    float add = 0.f; int cnt = 0;
    if (i < BP) {
        int c = conf_t[i];
        float v = ce[i];
        if (c > 0) { add = v; cnt = 1; }
        else if (c == 0) {
            int bin = (int)(__float_as_uint(v) >> 16);
            if (bin > sc->tb) { add = v; cnt = 1; }
            else if (bin == sc->tb) {
                int tk = atomicAdd(&sc->ticket, 1);
                if (tk < sc->rneed) { add = v; cnt = 1; }
            }
        }
    }
    if (cnt) { atomicAdd(&sSum, add); atomicAdd(&sCnt, 1); }
    __syncthreads();
    if (threadIdx.x == 0 && sCnt) {
        atomicAdd(&sc->keep_sum, sSum);
        atomicAdd(&sc->keep_cnt, sCnt);
    }
}

// ---------------- K6: finalize two scalar losses --------------------------
__global__ void k_final(const Scal* __restrict__ sc, float* __restrict__ out) {
    if (threadIdx.x == 0 && blockIdx.x == 0) {
        int np = sc->pos_count; if (np < 1) np = 1;
        out[0] = sc->pos_loss / (float)np;
        int kc = sc->keep_cnt; if (kc < 1) kc = 1;
        out[1] = sc->keep_sum / (float)kc;
    }
}

// ---------------- host launcher ------------------------------------------
extern "C" void kernel_launch(void* const* d_in, const int* in_sizes, int n_in,
                              void* d_out, int out_size, void* d_ws, size_t ws_size,
                              hipStream_t stream) {
    const float* loc    = (const float*)d_in[0];
    const float* conf   = (const float*)d_in[1];
    const float* priors = (const float*)d_in[2];
    const float* gt     = (const float*)d_in[3];
    const int*   labels = (const int*)d_in[4];

    int P  = in_sizes[2] / 4;
    int B  = in_sizes[0] / (4 * P);
    int BG = in_sizes[4];
    int G  = BG / B;
    int BP = B * P;
    int C  = in_sizes[1] / BP;

    char* ws = (char*)d_ws;
    Scal* sc     = (Scal*)ws;
    int*  hist   = (int*)(ws + 256);
    int*  conf_t = (int*)(ws + 256 + (size_t)NBINS * 4);
    float* ce    = (float*)(ws + 256 + (size_t)NBINS * 4 + (size_t)BP * 4);
    int*  bpi    = (int*)(ws + 256 + (size_t)NBINS * 4 + (size_t)BP * 8);

    k_init<<<(NBINS + 255) / 256, 256, 0, stream>>>(hist, sc);
    k_best_prior<<<BG, 256, 0, stream>>>(gt, priors, P, bpi);
    int bpb = (P + 255) / 256;
    k_match<<<B * bpb, 256, 0, stream>>>(loc, gt, labels, priors, bpi, G, P, conf_t, sc);
    int tiles = (BP + 15) / 16;
    k_ce<<<(tiles + 7) / 8, 256, 0, stream>>>(conf, conf_t, BP, C, ce, hist);
    k_scan<<<1, 256, 0, stream>>>(hist, sc);
    k_select<<<(BP + 255) / 256, 256, 0, stream>>>(conf_t, ce, BP, sc);
    k_final<<<1, 32, 0, stream>>>(sc, (float*)d_out);
}
